// CF_GEN_89670327206623
// MI455X (gfx1250) — matrix operand order repaired and checked by reading, not compiled, not run on hardware
//
#include <hip/hip_runtime.h>

// ---------------- problem constants (from reference) ----------------
#define BB   256
#define SS   50
#define KK   8
#define DD   512
#define NHH  8
#define HDD  64
#define NPASS 3
#define MTOK (NPASS * BB * SS)   // 38400 tokens, divisible by 32

typedef __bf16 bf16_t;
typedef __attribute__((ext_vector_type(16))) __bf16 v16bf;
typedef __attribute__((ext_vector_type(8)))  float  v8f;
typedef __attribute__((ext_vector_type(8)))  unsigned short v8us;

union BF16Frag { v16bf v; v8us h[2]; unsigned short u[16]; };

__device__ __forceinline__ unsigned short f2bf(float x) {
  unsigned int u = __float_as_uint(x);
  u += 0x7FFFu + ((u >> 16) & 1u);   // round-to-nearest-even
  return (unsigned short)(u >> 16);
}

// ---------------------------------------------------------------
// Pack a row-major f32 weight [K,N] into WMMA B-fragment layout:
// fragment fi = kt*(N/16)+nt holds 512 bf16: lane l, elem e ->
//   k = kt*32 + (l>>4)*8 + (e&7) + (e>>3)*16,  n = nt*16 + (l&15)
// ---------------------------------------------------------------
__global__ void __launch_bounds__(256) kpack(const float* __restrict__ W,
                                             unsigned short* __restrict__ Wp,
                                             int Kd, int N) {
  int idx = blockIdx.x * 256 + threadIdx.x;
  if (idx >= Kd * N) return;
  int fi  = idx >> 9;
  int rem = idx & 511;
  int l = rem >> 4, e = rem & 15;
  int NT = N >> 4;
  int kt = fi / NT, nt = fi - kt * NT;
  int k = kt * 32 + ((l >> 4) << 3) + (e & 7) + ((e >> 3) << 4);
  int n = (nt << 4) + (l & 15);
  Wp[idx] = f2bf(W[(size_t)k * N + n]);
}

// ---------------------------------------------------------------
// WMMA GEMM: Y[M,N] = act(X16[M,Kd] @ W + bias)
// A: bf16 row-major (pure 128-bit loads, no conversion VALU in loop).
// One wave computes a 32x64 strip: 8 accumulators, 2 A frags / k-step,
// 4 B frags shared between the two M-halves -> 8 WMMAs per iteration.
// Outputs: Yf (f32, optional) and/or Yh (bf16 row-major, optional).
// ---------------------------------------------------------------
__global__ void __launch_bounds__(256) kgemm(const unsigned short* __restrict__ X16,
                                             const unsigned short* __restrict__ Wp,
                                             const float* __restrict__ bias,
                                             float* __restrict__ Yf,
                                             unsigned short* __restrict__ Yh,
                                             int M, int Kd, int N, int relu) {
  const int wave = blockIdx.x * (blockDim.x >> 5) + (threadIdx.x >> 5);
  const int lane = threadIdx.x & 31;
  const int nt4  = N >> 6;
  const int mt   = wave / nt4;
  const int ng   = wave - mt * nt4;
  if (mt >= (M >> 5)) return;
  const int NT  = N >> 4;
  const int khi = (lane >> 4) << 3;          // 0 or 8 (A k-offset / C row-offset)

  v8f acc[8] = {};
  const unsigned short* x0 = X16 + (size_t)((mt << 5) + (lane & 15)) * Kd;
  const unsigned short* x1 = x0 + (size_t)16 * Kd;

  for (int k0 = 0; k0 < Kd; k0 += 32) {
    BF16Frag a0, a1;
    a0.h[0] = *(const v8us*)(x0 + k0 + khi);
    a0.h[1] = *(const v8us*)(x0 + k0 + khi + 16);
    a1.h[0] = *(const v8us*)(x1 + k0 + khi);
    a1.h[1] = *(const v8us*)(x1 + k0 + khi + 16);
    const unsigned short* wb =
        Wp + ((size_t)(k0 >> 5) * NT + ((size_t)ng << 2)) * 512 + ((size_t)lane << 4);
#pragma unroll
    for (int t = 0; t < 4; ++t) {
      BF16Frag bfr;
      bfr.v = *(const v16bf*)(wb + t * 512);
      acc[t] = __builtin_amdgcn_wmma_f32_16x16x32_bf16(
          false, a0.v, false, bfr.v, (short)0, acc[t], false, false);
      acc[4 + t] = __builtin_amdgcn_wmma_f32_16x16x32_bf16(
          false, a1.v, false, bfr.v, (short)0, acc[4 + t], false, false);
    }
  }

  // ---- epilogue: C/D layout row = base + khi + r, col = base + lane&15 ----
  const int col0 = (ng << 6) + (lane & 15);
#pragma unroll
  for (int half = 0; half < 2; ++half) {
    const int row0 = (mt << 5) + (half << 4) + khi;
#pragma unroll
    for (int t = 0; t < 4; ++t) {
      const int col = col0 + (t << 4);
      const float bv = bias[col];
      const v8f a = acc[half * 4 + t];
#pragma unroll
      for (int r = 0; r < 8; ++r) {
        float v = a[r] + bv;
        if (relu) v = v > 0.0f ? v : 0.0f;
        if (Yf) Yf[(size_t)(row0 + r) * N + col] = v;
        if (Yh) Yh[(size_t)(row0 + r) * N + col] = f2bf(v);
      }
    }
  }
}

// ---------------------------------------------------------------
// Embedding: info = sum_k knows_tab[..] + probs_tab[..] + pos_tab;
// x3 per pass adds types_tab[t]; f32 streams + bf16 GEMM mirrors.
// ---------------------------------------------------------------
__global__ void __launch_bounds__(256) kembed(const int* __restrict__ probs,
                                              const int* __restrict__ knows,
                                              const unsigned char* __restrict__ corrs,
                                              const float* __restrict__ ktab,
                                              const float* __restrict__ ptab,
                                              const float* __restrict__ ttab,
                                              const float* __restrict__ postab,
                                              float* __restrict__ info,
                                              float* __restrict__ xf,
                                              float* __restrict__ xb,
                                              unsigned short* __restrict__ xf16,
                                              unsigned short* __restrict__ xb16) {
  size_t i = (size_t)blockIdx.x * 256 + threadIdx.x;
  if (i >= (size_t)BB * SS * DD) return;
  const int d = (int)(i & (DD - 1));
  const size_t bs = i >> 9;
  const int s = (int)(bs % SS);
  float v = ptab[(size_t)probs[bs] * DD + d] + postab[(size_t)s * DD + d];
  const int* kr = knows + bs * KK;
#pragma unroll
  for (int j = 0; j < KK; ++j) v += ktab[(size_t)kr[j] * DD + d];
  info[i] = v;
  const bool c = corrs[bs] != 0;
  const int rt = c ? 1 : 2, wt = c ? 2 : 0, nt = c ? 1 : 0;
  const float e0 = v + ttab[rt * DD + d];
  const float e1 = v + ttab[wt * DD + d];
  const float e2 = v + ttab[nt * DD + d];
  const size_t P1 = (size_t)BB * SS * DD;
  xf[i] = e0; xf[i + P1] = e1; xf[i + 2 * P1] = e2;
  xb[i] = e0; xb[i + P1] = e1; xb[i + 2 * P1] = e2;
  const unsigned short h0 = f2bf(e0), h1 = f2bf(e1), h2 = f2bf(e2);
  xf16[i] = h0; xf16[i + P1] = h1; xf16[i + 2 * P1] = h2;
  xb16[i] = h0; xb16[i + P1] = h1; xb16[i + 2 * P1] = h2;
}

// ---------------------------------------------------------------
// Attention per (pass*B + b, head): scores -> mask -> softmax -> P@V
// backward=0: causal (block j>q); backward=1: anti-causal (block j<q).
// Key-padding mask from probs_his==0 (OR of query/key pads).
// ---------------------------------------------------------------
__global__ void __launch_bounds__(64) kattn(const float* __restrict__ Q,
                                            const float* __restrict__ Kin,
                                            const float* __restrict__ Vin,
                                            const int* __restrict__ probs,
                                            float* __restrict__ O, int backward) {
  __shared__ float Qs[SS][HDD + 1];
  __shared__ float Ks[SS][HDD + 1];
  __shared__ float Vs[SS][HDD + 1];
  __shared__ float Ps[SS][SS + 1];
  __shared__ unsigned char pad[SS];

  const int bh = blockIdx.x;
  const int h  = bh & (NHH - 1);
  const int bb = bh >> 3;
  const int b  = bb % BB;
  const size_t base = ((size_t)bb * SS) * DD + (size_t)h * HDD;
  const int tid = threadIdx.x;

  for (int idx = tid; idx < SS * HDD; idx += 64) {
    int j = idx >> 6, d = idx & 63;
    Qs[j][d] = Q[base + (size_t)j * DD + d];
    Ks[j][d] = Kin[base + (size_t)j * DD + d];
    Vs[j][d] = Vin[base + (size_t)j * DD + d];
  }
  if (tid < SS) pad[tid] = (probs[b * SS + tid] == 0) ? 1 : 0;
  __syncthreads();
  if (tid >= SS) return;
  const int q = tid;
  float* orow = O + base + (size_t)q * DD;
  if (pad[q]) {
    for (int d = 0; d < HDD; ++d) orow[d] = 0.0f;
    return;
  }
  float mx = -3.0e38f;
  for (int j = 0; j < SS; ++j) {
    const bool m = pad[j] || (backward ? (j < q) : (j > q));
    float a = -3.0e38f;
    if (!m) {
      float acc = 0.0f;
      for (int d = 0; d < HDD; ++d) acc += Qs[q][d] * Ks[j][d];
      a = acc * 0.125f;                 // HD^-0.5 = 1/8
      mx = fmaxf(mx, a);
    }
    Ps[q][j] = a;
  }
  float den = 0.0f;
  for (int j = 0; j < SS; ++j) {
    float a = Ps[q][j];
    float e = (a > -1.0e37f) ? __expf(a - mx) : 0.0f;
    Ps[q][j] = e;
    den += e;
  }
  const float inv = 1.0f / den;
  for (int d = 0; d < HDD; ++d) {
    float o = 0.0f;
    for (int j = 0; j < SS; ++j) o += Ps[q][j] * Vs[j][d];
    orow[d] = o * inv;
  }
}

// ---------------------------------------------------------------
// In-place x = LayerNorm(x + o) * g + be; also refresh bf16 mirror.
// ---------------------------------------------------------------
__global__ void __launch_bounds__(256) kaddln(float* __restrict__ x,
                                              const float* __restrict__ o,
                                              const float* __restrict__ g,
                                              const float* __restrict__ be,
                                              unsigned short* __restrict__ x16) {
  __shared__ float red[256];
  const int t = blockIdx.x, tid = threadIdx.x;
  float* xr = x + (size_t)t * DD;
  const float* orr = o + (size_t)t * DD;
  const float v0 = xr[tid] + orr[tid];
  const float v1 = xr[tid + 256] + orr[tid + 256];
  red[tid] = v0 + v1;
  __syncthreads();
  for (int st = 128; st > 0; st >>= 1) {
    if (tid < st) red[tid] += red[tid + st];
    __syncthreads();
  }
  const float mu = red[0] * (1.0f / DD);
  __syncthreads();
  const float d0 = v0 - mu, d1 = v1 - mu;
  red[tid] = d0 * d0 + d1 * d1;
  __syncthreads();
  for (int st = 128; st > 0; st >>= 1) {
    if (tid < st) red[tid] += red[tid + st];
    __syncthreads();
  }
  const float inv = rsqrtf(red[0] * (1.0f / DD) + 1e-5f);
  const float y0 = d0 * inv * g[tid] + be[tid];
  const float y1 = d1 * inv * g[tid + 256] + be[tid + 256];
  xr[tid] = y0;
  xr[tid + 256] = y1;
  unsigned short* hr = x16 + (size_t)t * DD;
  hr[tid] = f2bf(y0);
  hr[tid + 256] = f2bf(y1);
}

// ---------------------------------------------------------------
// out = cat(b[:,0], f[:,:-2]+b[:,2:], f[:,-1]); hcat16 = [out, info] (bf16)
// ---------------------------------------------------------------
__global__ void __launch_bounds__(256) kcombine(const float* __restrict__ f,
                                                const float* __restrict__ bk,
                                                const float* __restrict__ info,
                                                unsigned short* __restrict__ hcat) {
  size_t i = (size_t)blockIdx.x * 256 + threadIdx.x;
  if (i >= (size_t)MTOK * DD) return;
  const int d = (int)(i & (DD - 1));
  const size_t t = i >> 9;
  const int s = (int)(t % SS);
  const size_t bb = t / SS;
  const int b = (int)(bb % BB);
  float ov;
  if (s == 0)            ov = bk[i];
  else if (s == SS - 1)  ov = f[i];
  else                   ov = f[i - DD] + bk[i + DD];
  hcat[t * (2 * DD) + d]      = f2bf(ov);
  hcat[t * (2 * DD) + DD + d] = f2bf(info[((size_t)b * SS + s) * DD + d]);
}

// ---------------------------------------------------------------
// Final head: sigmoid(h2 @ pW3 + pb3), zeroed at padded positions.
// ---------------------------------------------------------------
__global__ void __launch_bounds__(256) khead(const float* __restrict__ h2,
                                             const float* __restrict__ w3,
                                             const float* __restrict__ b3,
                                             const int* __restrict__ probs,
                                             float* __restrict__ out) {
  const int t = blockIdx.x * 256 + threadIdx.x;
  if (t >= MTOK) return;
  const int s = t % SS;
  const int bb = t / SS;
  const int b = bb % BB;
  float z = b3[0];
  const float* hr = h2 + (size_t)t * (DD / 2);
  for (int i = 0; i < DD / 2; ++i) z += hr[i] * w3[i];
  const float sig = 1.0f / (1.0f + __expf(-z));
  out[t] = (probs[b * SS + s] == 0) ? 0.0f : sig;
}

// ---------------------------------------------------------------
extern "C" void kernel_launch(void* const* d_in, const int* in_sizes, int n_in,
                              void* d_out, int out_size, void* d_ws, size_t ws_size,
                              hipStream_t stream) {
  (void)in_sizes; (void)n_in; (void)out_size; (void)ws_size;
  const int*   probs  = (const int*)d_in[0];
  const int*   knows  = (const int*)d_in[1];
  const unsigned char* corrs = (const unsigned char*)d_in[2];
  const float* ktab   = (const float*)d_in[3];
  const float* ptab   = (const float*)d_in[4];
  const float* ttab   = (const float*)d_in[5];
  const float* postab = (const float*)d_in[6];
  const float* Wq  = (const float*)d_in[7];
  const float* Wk  = (const float*)d_in[8];
  const float* Wv  = (const float*)d_in[9];
  const float* W1  = (const float*)d_in[10];
  const float* W2  = (const float*)d_in[11];
  const float* bq  = (const float*)d_in[12];
  const float* bk  = (const float*)d_in[13];
  const float* bv  = (const float*)d_in[14];
  const float* b1  = (const float*)d_in[15];
  const float* b2  = (const float*)d_in[16];
  const float* be1 = (const float*)d_in[17];
  const float* be2 = (const float*)d_in[18];
  const float* g1  = (const float*)d_in[19];
  const float* g2  = (const float*)d_in[20];
  const float* pW1 = (const float*)d_in[21];
  const float* pb1 = (const float*)d_in[22];
  const float* pW2 = (const float*)d_in[23];
  const float* pb2 = (const float*)d_in[24];
  const float* pW3 = (const float*)d_in[25];
  const float* pb3 = (const float*)d_in[26];

  // ---- workspace carving ----
  char* ws = (char*)d_ws;
  size_t off = 0;
  auto carve = [&](size_t bytes) -> void* {
    void* p = ws + off;
    off += (bytes + 255) & ~(size_t)255;
    return p;
  };
  const size_t actF = (size_t)MTOK * DD * sizeof(float);          // 78.6 MB
  const size_t actH = (size_t)MTOK * DD * sizeof(unsigned short); // 39.3 MB
  float* info = (float*)carve((size_t)BB * SS * DD * sizeof(float));
  float* xf   = (float*)carve(actF);
  float* xb   = (float*)carve(actF);
  unsigned short* xf16 = (unsigned short*)carve(actH);
  unsigned short* xb16 = (unsigned short*)carve(actH);
  float* Qb   = (float*)carve(actF);   // also hcat16 overlay after blocks
  float* Kb2  = (float*)carve(actF);
  float* Vb2  = (float*)carve(actF);
  float* t0   = (float*)carve(actF);   // also h2 overlay after blocks
  unsigned short* h1b = (unsigned short*)carve(actH);  // FFN / pred hidden (bf16)
  unsigned short* WqP = (unsigned short*)carve(4ull * DD * DD * 2);
  unsigned short* WkP = (unsigned short*)carve(4ull * DD * DD * 2);
  unsigned short* WvP = (unsigned short*)carve(4ull * DD * DD * 2);
  unsigned short* W1P = (unsigned short*)carve(4ull * DD * DD * 2);
  unsigned short* W2P = (unsigned short*)carve(4ull * DD * DD * 2);
  unsigned short* pW1P = (unsigned short*)carve(2ull * DD * DD * 2);
  unsigned short* pW2P = (unsigned short*)carve((size_t)DD * (DD / 2) * 2);
  unsigned short* hcat16 = (unsigned short*)Qb;  // [MTOK,1024] bf16 = 78.6 MB
  float* h2 = t0;                                 // [MTOK,256] f32

  // ---- pack all weights to bf16 WMMA fragments ----
  for (int li = 0; li < 4; ++li) {
    const size_t mo = (size_t)li * DD * DD;
    kpack<<<(DD * DD) / 256, 256, 0, stream>>>(Wq + mo, WqP + mo, DD, DD);
    kpack<<<(DD * DD) / 256, 256, 0, stream>>>(Wk + mo, WkP + mo, DD, DD);
    kpack<<<(DD * DD) / 256, 256, 0, stream>>>(Wv + mo, WvP + mo, DD, DD);
    kpack<<<(DD * DD) / 256, 256, 0, stream>>>(W1 + mo, W1P + mo, DD, DD);
    kpack<<<(DD * DD) / 256, 256, 0, stream>>>(W2 + mo, W2P + mo, DD, DD);
  }
  kpack<<<(2 * DD * DD) / 256, 256, 0, stream>>>(pW1, pW1P, 2 * DD, DD);
  kpack<<<(DD * (DD / 2)) / 256, 256, 0, stream>>>(pW2, pW2P, DD, DD / 2);

  // ---- embeddings -> info, xf/xb (+ bf16 mirrors) ----
  kembed<<<(BB * SS * DD) / 256, 256, 0, stream>>>(probs, knows, corrs, ktab, ptab,
                                                   ttab, postab, info, xf, xb, xf16, xb16);

  // ---- transformer blocks ----
  const int g512 = (MTOK / 32) * (DD / 64) / 8;        // 1200 blocks (8 waves each)
  const int g256 = (MTOK / 32) * ((DD / 2) / 64) / 8;  // 600
  auto run_block = [&](float* X, unsigned short* X16, int li, int backward) {
    const size_t mo = (size_t)li * DD * DD;
    kgemm<<<g512, 256, 0, stream>>>(X16, WqP + mo, bq + li * DD, Qb, nullptr, MTOK, DD, DD, 0);
    kgemm<<<g512, 256, 0, stream>>>(X16, WkP + mo, bk + li * DD, Kb2, nullptr, MTOK, DD, DD, 0);
    kgemm<<<g512, 256, 0, stream>>>(X16, WvP + mo, bv + li * DD, Vb2, nullptr, MTOK, DD, DD, 0);
    kattn<<<NPASS * BB * NHH, 64, 0, stream>>>(Qb, Kb2, Vb2, probs, t0, backward);
    kaddln<<<MTOK, 256, 0, stream>>>(X, t0, g1 + li * DD, be1 + li * DD, X16);
    kgemm<<<g512, 256, 0, stream>>>(X16, W1P + mo, b1 + li * DD, nullptr, h1b, MTOK, DD, DD, 1);
    kgemm<<<g512, 256, 0, stream>>>(h1b, W2P + mo, b2 + li * DD, Qb, nullptr, MTOK, DD, DD, 0);
    kaddln<<<MTOK, 256, 0, stream>>>(X, Qb, g2 + li * DD, be2 + li * DD, X16);
  };
  run_block(xf, xf16, 0, 0);
  run_block(xf, xf16, 1, 0);
  run_block(xb, xb16, 2, 1);
  run_block(xb, xb16, 3, 1);

  // ---- combine + prediction MLP ----
  kcombine<<<(int)(((size_t)MTOK * DD) / 256), 256, 0, stream>>>(xf, xb, info, hcat16);
  kgemm<<<g512, 256, 0, stream>>>(hcat16, pW1P, pb1, nullptr, h1b, MTOK, 2 * DD, DD, 1);
  kgemm<<<g256, 256, 0, stream>>>(h1b, pW2P, pb2, h2, nullptr, MTOK, DD, DD / 2, 1);
  khead<<<(MTOK + 255) / 256, 256, 0, stream>>>(h2, pW3, pb3, probs, (float*)d_out);
}
